// Sage2_20315195310685
// MI455X (gfx1250) — matrix operand order repaired and checked by reading, not compiled, not run on hardware
//
#include <hip/hip_runtime.h>

#define N_NODES 100000
#define N_EDGES 1600000
#define NUM_GRAPHS 1024
#define D 128

typedef __attribute__((ext_vector_type(16))) unsigned short u16x16;
typedef __attribute__((ext_vector_type(8)))  unsigned short u16x8;
typedef __attribute__((ext_vector_type(16))) __bf16 bf16x16;
typedef __attribute__((ext_vector_type(8)))  float  f32x8;

__device__ __forceinline__ unsigned short f2bf(float f) {
    unsigned int u = __builtin_bit_cast(unsigned int, f);
    u += 0x7FFFu + ((u >> 16) & 1u);   // round-to-nearest-even
    return (unsigned short)(u >> 16);
}
__device__ __forceinline__ float bf2f(unsigned short h) {
    unsigned int u = ((unsigned int)h) << 16;
    return __builtin_bit_cast(float, u);
}

// ---------------------------------------------------------------- degree count
__global__ void deg_kernel(const int* __restrict__ dst, float* __restrict__ deg) {
    int e = blockIdx.x * blockDim.x + threadIdx.x;
    if (e < N_EDGES) atomicAdd(&deg[dst[e]], 1.0f);
}

// -------------------------------------------- one-time f32 -> bf16 row-major cvt
// 8 elements per thread: two float4 loads -> one 16B store
__global__ void cvt_bf16_kernel(const float* __restrict__ x, unsigned short* __restrict__ xb,
                                long long n8) {
    long long t = (long long)blockIdx.x * blockDim.x + threadIdx.x;
    if (t >= n8) return;
    float4 a = ((const float4*)x)[t * 2 + 0];
    float4 b = ((const float4*)x)[t * 2 + 1];
    u16x8 o;
    o[0] = f2bf(a.x); o[1] = f2bf(a.y); o[2] = f2bf(a.z); o[3] = f2bf(a.w);
    o[4] = f2bf(b.x); o[5] = f2bf(b.y); o[6] = f2bf(b.z); o[7] = f2bf(b.w);
    ((u16x8*)xb)[t] = o;
}

// ------------------------------- one-time weight pack into WMMA B-operand layout
// B tile (nt, kb): lane l supplies col = nt*16 + (l&15), K = kb*32 + (l>>4)*8 + (i<8 ? i : i+8)
// (B[k][n] = W[n][k], W row-major [out,in] -> two runs of 8 floats per lane)
// Packed flat: (((nt*4 + kb)*32 + lane)*16 + i) bf16 values, 32B per lane.
__global__ void pack_w_kernel(const float* __restrict__ W, unsigned short* __restrict__ out) {
    int t = blockIdx.x * blockDim.x + threadIdx.x;  // 0..1023
    if (t >= 1024) return;
    int lane = t & 31;
    int kb = (t >> 5) & 3;
    int nt = t >> 7;
    int col  = nt * 16 + (lane & 15);
    int koff = kb * 32 + (lane >> 4) * 8;
    const float* wp = W + (size_t)col * D + koff;
    u16x16 o;
#pragma unroll
    for (int i = 0; i < 16; ++i) o[i] = f2bf(wp[i < 8 ? i : i + 8]);
    ((u16x16*)out)[t] = o;
}

// ------------------------------------------------ edge scatter (bf16 features)
// 32 threads per edge, 4 bf16 per thread (8B gather, 4 f32 atomic adds)
__global__ void scatter_bf16_kernel(const int* __restrict__ src, const int* __restrict__ dst,
                                    const unsigned short* __restrict__ h, float* __restrict__ msg) {
    long long t = (long long)blockIdx.x * blockDim.x + threadIdx.x;
    long long e = t >> 5;
    if (e >= N_EDGES) return;
    int c = (int)(t & 31);
    int s = src[e], d = dst[e];
    uint2 r = *(const uint2*)(h + (size_t)s * D + c * 4);
    float* p = msg + (size_t)d * D + c * 4;
    atomicAdd(p + 0, bf2f((unsigned short)(r.x & 0xFFFFu)));
    atomicAdd(p + 1, bf2f((unsigned short)(r.x >> 16)));
    atomicAdd(p + 2, bf2f((unsigned short)(r.y & 0xFFFFu)));
    atomicAdd(p + 3, bf2f((unsigned short)(r.y >> 16)));
}

// ------------------------------------------------------------- fused SAGE GEMM
// out = relu( (msg/deg) @ Wl^T + bl + in @ Wr^T )
// One wave per 16-node row tile; 8 N-tiles of 16; K=128 in 4 steps of 32.
// hin: bf16 row-major input (xb for layer1, h1 for layer2).
// Wlp/Wrp: pre-packed bf16 B operands (zero conversion work in here).
template <bool LAYER2>
__global__ void sage_wmma_kernel(const float* __restrict__ msg, const float* __restrict__ deg,
                                 const unsigned short* __restrict__ hin,
                                 const unsigned short* __restrict__ Wlp,
                                 const float* __restrict__ bl,
                                 const unsigned short* __restrict__ Wrp,
                                 unsigned short* __restrict__ hout,  // layer1 output (bf16)
                                 float* __restrict__ gpool,          // layer2 pooled output
                                 const int* __restrict__ batch) {
    const int NT = N_NODES / 16;  // 6250 row tiles (exact)
    int wid = blockIdx.x * (blockDim.x >> 5) + (threadIdx.x >> 5);
    if (wid >= NT) return;        // wave-uniform -> EXEC stays all-1s for WMMA
    int lane  = threadIdx.x & 31;
    int m0    = wid * 16;
    int lrow  = m0 + (lane & 15); // A-matrix row supplied by this lane
    int khalf = lane >> 4;

    float inv = 1.0f / fmaxf(deg[lrow], 1.0f);
    const float* msg_row = msg + (size_t)lrow * D;
    const unsigned short* hr = hin + (size_t)lrow * D;

    // A operands for all 4 K-blocks, reused across the 8 N-tiles.
    // 16-bit A 16x32 layout: lanes 0-15 K {+0..7,+16..23}; lanes 16-31 K {+8..15,+24..31}
    bf16x16 Aagg[4], Ain[4];
#pragma unroll
    for (int kb = 0; kb < 4; ++kb) {
        int ab = kb * 32 + khalf * 8;
        bf16x16 a;
#pragma unroll
        for (int i = 0; i < 8; ++i) {
            a[i]     = (__bf16)(msg_row[ab + i]      * inv);
            a[8 + i] = (__bf16)(msg_row[ab + 16 + i] * inv);
        }
        Aagg[kb] = a;
        u16x8 lo = *(const u16x8*)(hr + ab);        // raw bf16, no conversion
        u16x8 hi = *(const u16x8*)(hr + ab + 16);
        u16x16 ui;
#pragma unroll
        for (int i = 0; i < 8; ++i) { ui[i] = lo[i]; ui[8 + i] = hi[i]; }
        Ain[kb] = __builtin_bit_cast(bf16x16, ui);
    }

    const u16x16* blp = (const u16x16*)Wlp;
    const u16x16* brp = (const u16x16*)Wrp;

    for (int nt = 0; nt < 8; ++nt) {
        int col = nt * 16 + (lane & 15);
        f32x8 acc = {};
#pragma unroll
        for (int kb = 0; kb < 4; ++kb) {
            int pidx = (nt * 4 + kb) * 32 + lane;
            bf16x16 Bl = __builtin_bit_cast(bf16x16, blp[pidx]);  // 2x b128 load
            bf16x16 Br = __builtin_bit_cast(bf16x16, brp[pidx]);
            acc = __builtin_amdgcn_wmma_f32_16x16x32_bf16(false, Aagg[kb], false, Bl,
                                                          (short)0, acc, false, false);
            acc = __builtin_amdgcn_wmma_f32_16x16x32_bf16(false, Ain[kb],  false, Br,
                                                          (short)0, acc, false, false);
        }
        float bias = bl[col];
        // C/D 16x16 f32 layout: VGPR v -> M = khalf*8 + v; N = lane&15
#pragma unroll
        for (int v = 0; v < 8; ++v) {
            int row = m0 + khalf * 8 + v;
            float val = fmaxf(acc[v] + bias, 0.0f);
            if (LAYER2) {
                atomicAdd(&gpool[(size_t)batch[row] * D + col], val);  // fused add-pool
            } else {
                hout[(size_t)row * D + col] = f2bf(val);
            }
        }
    }
}

// ------------------------------------------- LayerNorm + 128->2 linear, 1 wave/graph
__global__ void ln_lin_kernel(const float* __restrict__ g, const float* __restrict__ ln_g,
                              const float* __restrict__ ln_b, const float* __restrict__ Wlin,
                              const float* __restrict__ blin, float* __restrict__ out) {
    int wid = blockIdx.x * (blockDim.x >> 5) + (threadIdx.x >> 5);
    if (wid >= NUM_GRAPHS) return;
    int lane = threadIdx.x & 31;
    float4 v = ((const float4*)(g + (size_t)wid * D))[lane];
    float s = v.x + v.y + v.z + v.w;
    for (int m = 16; m >= 1; m >>= 1) s += __shfl_xor(s, m, 32);
    float mean = s * (1.0f / D);
    float d0 = v.x - mean, d1 = v.y - mean, d2 = v.z - mean, d3 = v.w - mean;
    float q = d0 * d0 + d1 * d1 + d2 * d2 + d3 * d3;
    for (int m = 16; m >= 1; m >>= 1) q += __shfl_xor(q, m, 32);
    float rstd = rsqrtf(q * (1.0f / D) + 1e-5f);
    int j = lane * 4;
    float g0 = d0 * rstd * ln_g[j + 0] + ln_b[j + 0];
    float g1 = d1 * rstd * ln_g[j + 1] + ln_b[j + 1];
    float g2 = d2 * rstd * ln_g[j + 2] + ln_b[j + 2];
    float g3 = d3 * rstd * ln_g[j + 3] + ln_b[j + 3];
    float o0 = g0 * Wlin[j + 0] + g1 * Wlin[j + 1] + g2 * Wlin[j + 2] + g3 * Wlin[j + 3];
    float o1 = g0 * Wlin[D + j + 0] + g1 * Wlin[D + j + 1] +
               g2 * Wlin[D + j + 2] + g3 * Wlin[D + j + 3];
    for (int m = 16; m >= 1; m >>= 1) { o0 += __shfl_xor(o0, m, 32); o1 += __shfl_xor(o1, m, 32); }
    if (lane == 0) {
        out[wid * 2 + 0] = o0 + blin[0];
        out[wid * 2 + 1] = o1 + blin[1];
    }
}

extern "C" void kernel_launch(void* const* d_in, const int* in_sizes, int n_in,
                              void* d_out, int out_size, void* d_ws, size_t ws_size,
                              hipStream_t stream) {
    (void)in_sizes; (void)n_in; (void)out_size; (void)ws_size;
    const float* x     = (const float*)d_in[0];
    const int*   ei    = (const int*)d_in[1];
    const int*   batch = (const int*)d_in[2];
    const float* W1l   = (const float*)d_in[3];
    const float* b1l   = (const float*)d_in[4];
    const float* W1r   = (const float*)d_in[5];
    const float* W2l   = (const float*)d_in[6];
    const float* b2l   = (const float*)d_in[7];
    const float* W2r   = (const float*)d_in[8];
    const float* ln_g  = (const float*)d_in[9];
    const float* ln_b  = (const float*)d_in[10];
    const float* Wlin  = (const float*)d_in[11];
    const float* blin  = (const float*)d_in[12];
    const int* src = ei;
    const int* dst = ei + N_EDGES;

    char* ws = (char*)d_ws;
    size_t off = 0;
    float* msg = (float*)(ws + off); off += (size_t)N_NODES * D * sizeof(float);    // 51.2 MB
    float* deg = (float*)(ws + off); off += (size_t)N_NODES * sizeof(float);        //  0.4 MB
    float* g   = (float*)(ws + off); off += (size_t)NUM_GRAPHS * D * sizeof(float); //  0.5 MB
    unsigned short* h1 = (unsigned short*)(ws + off); off += (size_t)N_NODES * D * 2; // 25.6 MB
    unsigned short* xb = (unsigned short*)(ws + off); off += (size_t)N_NODES * D * 2; // 25.6 MB
    unsigned short* p1l = (unsigned short*)(ws + off); off += (size_t)D * D * 2;    // 32 KB
    unsigned short* p1r = (unsigned short*)(ws + off); off += (size_t)D * D * 2;
    unsigned short* p2l = (unsigned short*)(ws + off); off += (size_t)D * D * 2;
    unsigned short* p2r = (unsigned short*)(ws + off); off += (size_t)D * D * 2;

    hipMemsetAsync(msg, 0, (size_t)N_NODES * D * sizeof(float), stream);
    hipMemsetAsync(deg, 0, (size_t)N_NODES * sizeof(float), stream);
    hipMemsetAsync(g,   0, (size_t)NUM_GRAPHS * D * sizeof(float), stream);

    // one-time preconversions (trivial cost: ~77 MB of traffic total)
    pack_w_kernel<<<4, 256, 0, stream>>>(W1l, p1l);
    pack_w_kernel<<<4, 256, 0, stream>>>(W1r, p1r);
    pack_w_kernel<<<4, 256, 0, stream>>>(W2l, p2l);
    pack_w_kernel<<<4, 256, 0, stream>>>(W2r, p2r);
    long long n8 = (long long)N_NODES * D / 8;
    cvt_bf16_kernel<<<(int)((n8 + 255) / 256), 256, 0, stream>>>(x, xb, n8);

    deg_kernel<<<(N_EDGES + 255) / 256, 256, 0, stream>>>(dst, deg);

    long long sc_threads = (long long)N_EDGES * 32;
    int sc_blocks = (int)((sc_threads + 255) / 256);
    int rt_blocks = (N_NODES / 16 + 7) / 8;   // 8 waves / block

    // layer 1 (gather in bf16 from xb)
    scatter_bf16_kernel<<<sc_blocks, 256, 0, stream>>>(src, dst, xb, msg);
    sage_wmma_kernel<false><<<rt_blocks, 256, 0, stream>>>(msg, deg, xb, p1l, b1l, p1r,
                                                           h1, nullptr, nullptr);

    // layer 2 (gather in bf16 from h1, fused pooling epilogue)
    hipMemsetAsync(msg, 0, (size_t)N_NODES * D * sizeof(float), stream);
    scatter_bf16_kernel<<<sc_blocks, 256, 0, stream>>>(src, dst, h1, msg);
    sage_wmma_kernel<true><<<rt_blocks, 256, 0, stream>>>(msg, deg, h1, p2l, b2l, p2r,
                                                          nullptr, g, batch);

    ln_lin_kernel<<<(NUM_GRAPHS + 7) / 8, 256, 0, stream>>>(g, ln_g, ln_b, Wlin, blin,
                                                            (float*)d_out);
}
